// MolecularGraphNeuralNetwork_68195490726331
// MI455X (gfx1250) — matrix operand order repaired and checked by reading, not compiled, not run on hardware
//
#include <hip/hip_runtime.h>
#include <hip/hip_bf16.h>
#include <math.h>

// MolecularGraphNeuralNetwork fused kernel for gfx1250 (MI455X).
// One workgroup per molecule (128 blocks x 256 threads = 8 wave32).
// Block-diagonal adjacency exploited: only the 64x64 diagonal blocks are read
// (2 MB instead of 768 MB of dense-adjacency traffic).
// 64x64 matmuls run on v_wmma_f32_16x16x32_f16 (f16 A/B, f32 accumulate).

#define TPB   256
#define NMOL  64
#define DIM   64
#define MTOT  8192
#define NPROP 452
#define LDS_S 72   // row stride (floats) for 64-wide LDS tiles, dodges bank conflicts

typedef __attribute__((ext_vector_type(16))) _Float16 v16h;
typedef __attribute__((ext_vector_type(8)))  float    v8f;

// A-fragment (16x32, f16) per CDNA5 ISA layout:
// lanes 0-15: M=lane, K in {0..7, 16..23}; lanes 16-31: M=lane-16, K in {8..15, 24..31};
// VGPR p (p<4): K = kOff+2p+{0,1}; VGPR p (p>=4): K = 16+kOff+2(p-4)+{0,1}.
__device__ __forceinline__ v16h load_a_frag(const float* __restrict__ S, int ldS,
                                            int mBase, int kBase, int lane) {
  const int row  = mBase + (lane & 15);
  const int kOff = (lane < 16) ? 0 : 8;
  const float* rp = S + row * ldS + kBase;
  v16h a;
#pragma unroll
  for (int e = 0; e < 16; ++e) {
    const int p = e >> 1, h = e & 1;
    const int k = (p < 4) ? (kOff + 2 * p + h) : (16 + kOff + 2 * (p - 4) + h);
    a[e] = (_Float16)rp[k];
  }
  return a;
}

// B-fragment (32x16, f16): lane N = lane&15; lanes 0-15 hold K in {0..7, 16..23},
// lanes 16-31 hold K in {8..15, 24..31}.
__device__ __forceinline__ v16h load_b_frag(const float* __restrict__ W, int ldW,
                                            int nBase, int kBase, int lane) {
  const int col  = nBase + (lane & 15);
  const int kOff = (lane < 16) ? 0 : 8;
  const float* cp = W + (size_t)(kBase + kOff) * ldW + col;
  v16h b;
#pragma unroll
  for (int e = 0; e < 16; ++e) b[e] = (_Float16)cp[(size_t)(e < 8 ? e : e + 8) * ldW];
  return b;
}

// Dst(64x64,ldD) = X(64x64,ldX) @ W(64x64,ldW), optionally fused +bias,ReLU.
// 8 waves: wave>>1 picks the 16-row stripe, wave&1 picks which pair of 16-col tiles.
template <bool RELU_BIAS>
__device__ __forceinline__ void mm64(float* __restrict__ Dst, int ldD,
                                     const float* __restrict__ X, int ldX,
                                     const float* __restrict__ W, int ldW,
                                     const float* __restrict__ bias,
                                     int wave, int lane) {
  const int mBase = (wave >> 1) * 16;
  const v16h a0 = load_a_frag(X, ldX, mBase, 0,  lane);
  const v16h a1 = load_a_frag(X, ldX, mBase, 32, lane);
#pragma unroll
  for (int t = 0; t < 2; ++t) {
    const int nBase = (((wave & 1) << 1) + t) * 16;
    const v16h b0 = load_b_frag(W, ldW, nBase, 0,  lane);
    const v16h b1 = load_b_frag(W, ldW, nBase, 32, lane);
    v8f acc = {};
    acc = __builtin_amdgcn_wmma_f32_16x16x32_f16(false, a0, false, b0,
                                                 (short)0, acc, false, false);
    acc = __builtin_amdgcn_wmma_f32_16x16x32_f16(false, a1, false, b1,
                                                 (short)0, acc, false, false);
    // D layout: VGPR r -> row mBase + r + (lane<16 ? 0 : 8), col = nBase + (lane&15)
    const int col  = nBase + (lane & 15);
    const int rOff = mBase + ((lane < 16) ? 0 : 8);
#pragma unroll
    for (int r = 0; r < 8; ++r) {
      float v = acc[r];
      if constexpr (RELU_BIAS) {
        v += bias[col];
        v = v > 0.f ? v : 0.f;
      }
      Dst[(rOff + r) * ldD + col] = v;
    }
  }
}

__global__ __launch_bounds__(TPB) void MolecularGraphNeuralNetwork_68195490726331_kernel(
    const int*   __restrict__ fp,     // [8192]
    const float* __restrict__ adj,    // [8192,8192] block-diagonal
    const float* __restrict__ embed,  // [10000,64]
    const float* __restrict__ Wf,     // [3,64,64]
    const float* __restrict__ bf,     // [3,64]
    const float* __restrict__ Wo,     // [3,64,64]
    const float* __restrict__ bo,     // [3,64]
    const float* __restrict__ Wp,     // [64,452]
    const float* __restrict__ bp,     // [452]
    float*       __restrict__ out)    // [128,452]
{
  __shared__ float sV[NMOL * LDS_S];  // node features v / temp T = A@h
  __shared__ float sH[NMOL * LDS_S];  // h = relu(v@Wf+bf)
  __shared__ float sA[NMOL * LDS_S];  // 64x64 diagonal adjacency block
  __shared__ float sM0[DIM];
  __shared__ float sM1[DIM];

  const int tid  = threadIdx.x;
  const int lane = tid & 31;
  const int wave = tid >> 5;       // 0..7
  const int mol  = blockIdx.x;     // 0..127
  const int base = mol * NMOL;     // first node of this molecule

  // ---- stage: embed gather -> sV ; diagonal adjacency block -> sA (float4) ----
  for (int i = tid; i < NMOL * (DIM / 4); i += TPB) {
    const int r  = i >> 4;
    const int c4 = (i & 15) << 2;
    const float4 e4 = *(const float4*)(&embed[(size_t)fp[base + r] * DIM + c4]);
    *(float4*)(&sV[r * LDS_S + c4]) = e4;
    const float4 a4 = *(const float4*)(&adj[(size_t)(base + r) * MTOT + base + c4]);
    *(float4*)(&sA[r * LDS_S + c4]) = a4;
  }
  __syncthreads();

  // ---- 3 GNN layers, fully in LDS ----
  for (int l = 0; l < 3; ++l) {
    const float* Wl = Wf + (size_t)l * DIM * DIM;  // hot in L2, shared by all WGs
    const float* bl = bf + l * DIM;

    // h = relu(v @ Wf[l] + bf[l])
    mm64<true>(sH, LDS_S, sV, LDS_S, Wl, DIM, bl, wave, lane);
    __syncthreads();

    // T = A_block @ h  (stored into sV; old v is dead)
    mm64<false>(sV, LDS_S, sA, LDS_S, sH, LDS_S, nullptr, wave, lane);
    __syncthreads();

    // v = (h + T) / max(||h + T||_2, 1e-12), one thread per row
    if (tid < NMOL) {
      float s = 0.f;
#pragma unroll 8
      for (int c = 0; c < DIM; ++c) {
        const float x = sH[tid * LDS_S + c] + sV[tid * LDS_S + c];
        s += x * x;
      }
      float n = sqrtf(s);
      n = n > 1e-12f ? n : 1e-12f;
      const float inv = 1.f / n;
#pragma unroll 8
      for (int c = 0; c < DIM; ++c)
        sV[tid * LDS_S + c] = (sH[tid * LDS_S + c] + sV[tid * LDS_S + c]) * inv;
    }
    __syncthreads();
  }

  // ---- sum-pool over the molecule's 64 nodes ----
  if (tid < DIM) {
    float s = 0.f;
#pragma unroll 8
    for (int r = 0; r < NMOL; ++r) s += sV[r * LDS_S + tid];
    sM0[tid] = s;
  }
  __syncthreads();

  // ---- MLP head: 3 x relu(mol @ Wo + bo), ping-pong buffers ----
  float* mIn  = sM0;
  float* mOut = sM1;
  for (int l = 0; l < 3; ++l) {
    if (tid < DIM) {
      const float* W = Wo + (size_t)l * DIM * DIM;
      float s = bo[l * DIM + tid];
#pragma unroll 8
      for (int d = 0; d < DIM; ++d) s += mIn[d] * W[d * DIM + tid];
      mOut[tid] = s > 0.f ? s : 0.f;
    }
    __syncthreads();
    float* t = mIn; mIn = mOut; mOut = t;
  }

  // ---- final projection to 452 properties ----
  for (int j = tid; j < NPROP; j += TPB) {
    float s = bp[j];
#pragma unroll 8
    for (int d = 0; d < DIM; ++d) s += mIn[d] * Wp[(size_t)d * NPROP + j];
    out[(size_t)mol * NPROP + j] = s;
  }
}

extern "C" void kernel_launch(void* const* d_in, const int* in_sizes, int n_in,
                              void* d_out, int out_size, void* d_ws, size_t ws_size,
                              hipStream_t stream) {
  (void)in_sizes; (void)n_in; (void)d_ws; (void)ws_size; (void)out_size;
  const int*   fp    = (const int*)  d_in[0];
  const float* adj   = (const float*)d_in[1];
  // d_in[2] = mol_size scalar (== 64), baked in as a compile-time constant
  const float* embed = (const float*)d_in[3];
  const float* Wf    = (const float*)d_in[4];
  const float* bf    = (const float*)d_in[5];
  const float* Wo    = (const float*)d_in[6];
  const float* bo    = (const float*)d_in[7];
  const float* Wp    = (const float*)d_in[8];
  const float* bp    = (const float*)d_in[9];
  float* out = (float*)d_out;

  const int nMol = MTOT / NMOL;  // 128 workgroups, one molecule each
  MolecularGraphNeuralNetwork_68195490726331_kernel<<<nMol, TPB, 0, stream>>>(
      fp, adj, embed, Wf, bf, Wo, bo, Wp, bp, out);
}
